// Block_37220186587207
// MI455X (gfx1250) — matrix operand order repaired and checked by reading, not compiled, not run on hardware
//
#include <hip/hip_runtime.h>

// ---------------------------------------------------------------------------
// Transformer block (LN -> QKV -> causal flash attention -> proj+res -> LN ->
// FFN+res) for MI455X / gfx1250. All GEMM-shaped math runs on
// v_wmma_f32_16x16x32_bf16 (fp32 accumulate), wave32.
// Weights are pre-transposed to [N][K] bf16 once per call; GEMM K-loop is
// double-buffered. Tile staging uses the Tensor Data Mover
// (tensor_load_to_lds + s_wait_tensorcnt) when the builtin is available,
// falling back to register-staged vectorized copies otherwise.
// ---------------------------------------------------------------------------

typedef __attribute__((ext_vector_type(16))) __bf16 v16bf;
typedef __attribute__((ext_vector_type(8)))  __bf16 v8bf;
typedef __attribute__((ext_vector_type(8)))  float  v8f;
typedef __attribute__((ext_vector_type(4)))  unsigned int u32x4;
typedef __attribute__((ext_vector_type(4)))  int  i32x4;
typedef __attribute__((ext_vector_type(8)))  int  i32x8;

#define Bb   4
#define Ts   2048
#define Cc   1024
#define Hh   16
#define HSz  64
#define Mrows (Bb * Ts)   // 8192

#if __has_builtin(__builtin_amdgcn_tensor_load_to_lds)
#define GEMM_USE_TDM 1
#else
#define GEMM_USE_TDM 0
#endif

static __device__ __forceinline__ v8f v8f_zero() {
  v8f z;
#pragma unroll
  for (int i = 0; i < 8; ++i) z[i] = 0.0f;
  return z;
}

// A-fragment (16x32 bf16): lane holds row M = lane%16; element e maps to
// k = (e/8)*16 + 8*(lane/16) + (e%8)  ->  two contiguous 8-elem (16B) loads.
static __device__ __forceinline__ v16bf load_a_frag(const __bf16* row, int half) {
  v8bf lo = *(const v8bf*)(row + 8 * half);
  v8bf hi = *(const v8bf*)(row + 16 + 8 * half);
  v16bf a;
#pragma unroll
  for (int i = 0; i < 8; ++i) { a[i] = lo[i]; a[i + 8] = hi[i]; }
  return a;
}

// B-fragment (32x16 bf16): lane holds col N = lane%16; element e maps to
// k = (e/8)*16 + 8*(lane/16) + (e%8)  ->  two contiguous 8-elem (16B) loads.
static __device__ __forceinline__ v16bf load_b_frag(const __bf16* krow, int half) {
  v8bf lo = *(const v8bf*)(krow + 8 * half);
  v8bf hi = *(const v8bf*)(krow + 16 + 8 * half);
  v16bf b;
#pragma unroll
  for (int i = 0; i < 8; ++i) { b[i] = lo[i]; b[i + 8] = hi[i]; }
  return b;
}

static __device__ __forceinline__ v8f wmma_bf16(v16bf a, v16bf b, v8f c) {
  return __builtin_amdgcn_wmma_f32_16x16x32_bf16(
      /*neg_a=*/false, a, /*neg_b=*/false, b,
      /*c_mod=*/(short)0, c, /*reuse_a=*/false, /*reuse_b=*/false);
}

static __device__ __forceinline__ float rmax16(float v) {
#pragma unroll
  for (int m = 8; m >= 1; m >>= 1) v = fmaxf(v, __shfl_xor(v, m, 16));
  return v;
}
static __device__ __forceinline__ float rsum16(float v) {
#pragma unroll
  for (int m = 8; m >= 1; m >>= 1) v += __shfl_xor(v, m, 16);
  return v;
}

#if GEMM_USE_TDM
static __device__ __forceinline__ void wait_tensorcnt0() {
#if __has_builtin(__builtin_amdgcn_s_wait_tensorcnt)
  __builtin_amdgcn_s_wait_tensorcnt(0);
#else
  asm volatile("s_wait_tensorcnt 0x0" ::: "memory");
#endif
}

// TDM load of a 128-row x 64B tile (row stride = K bf16 elements) into LDS.
// D# per CDNA5 ISA: group0 = {count|flags, lds_addr,
// global_addr lo, global_addr hi | type=2}; group1 carries data_size (8B),
// tensor dims, tile dims (8 units x 128 rows) and dim0 stride (K/4 units).
// All inputs are workgroup-uniform -> descriptor stays in SGPRs.
static __device__ __forceinline__ void tdm_load_tile(const __bf16* gsrc,
                                                     unsigned ldsAddr, int K) {
  const unsigned long long ga = (unsigned long long)(uintptr_t)gsrc;
  const unsigned units = (unsigned)(K >> 2);   // row length in 8B units
  u32x4 g0;
  g0[0] = 1u;                                       // count=1, user mode
  g0[1] = ldsAddr;                                  // LDS byte address
  g0[2] = (unsigned)(ga & 0xFFFFFFFFu);             // global_addr[31:0]
  g0[3] = (unsigned)((ga >> 32) & 0x1FFFFFFu)       // global_addr[56:32]
          | (2u << 30);                             // type = 2 ("image")
  i32x8 g1;
  g1[0] = (3 << 16);                                // data_size = 8 bytes
  g1[1] = (int)(units << 16);                       // tensor_dim0[15:0]
  g1[2] = (int)((units >> 16) & 0xFFFFu)            // tensor_dim0[31:16]
          | (128 << 16);                            // tensor_dim1 = 128
  g1[3] = (8 << 16);                                // tile_dim0 = 8 units (64B)
  g1[4] = 128;                                      // tile_dim1 = 128, tile_dim2 = 0
  g1[5] = (int)units;                               // tensor_dim0_stride[31:0]
  g1[6] = 0;                                        // stride[47:32], dim1_stride lo
  g1[7] = 0;
  i32x4 z4; z4[0] = 0; z4[1] = 0; z4[2] = 0; z4[3] = 0;
#if __clang_major__ >= 23
  i32x8 z8;
#pragma unroll
  for (int i = 0; i < 8; ++i) z8[i] = 0;
  __builtin_amdgcn_tensor_load_to_lds(g0, g1, z4, z4, z8, 0);
#else
  __builtin_amdgcn_tensor_load_to_lds(g0, g1, z4, z4, 0);
#endif
}
#endif  // GEMM_USE_TDM

// ---------------------------------------------------------------------------
// Repack / convert kernels (run once per call; tiny vs. GEMM cost)
// ---------------------------------------------------------------------------

// Wq/Wk/Wv are (H, C, HS). Build transposed WqkvT as [3072][1024] bf16:
// row n = which*1024 + h*64 + d, col c.  (writes coalesced along c)
__global__ void pack_qkv_wT_kernel(const float* __restrict__ Wq,
                                   const float* __restrict__ Wk,
                                   const float* __restrict__ Wv,
                                   __bf16* __restrict__ out) {
  int i = blockIdx.x * 256 + threadIdx.x;
  if (i >= 3 * Cc * Cc) return;
  int n = i / Cc;                 // output row
  int c = i - n * Cc;             // output col = K index
  int which = n >> 10, hh = (n >> 6) & 15, dd = n & 63;
  const float* W = (which == 0) ? Wq : (which == 1) ? Wk : Wv;
  out[i] = (__bf16)W[((size_t)hh * Cc + c) * HSz + dd];
}

// Tiled fp32[K][N] -> bf16[N][K] transpose (coalesced both sides via LDS)
__global__ __launch_bounds__(256) void transpose_to_bf16(
    const float* __restrict__ in, __bf16* __restrict__ out, int K, int N) {
  __shared__ float tile[32][33];
  const int kb = blockIdx.y * 32, nb = blockIdx.x * 32;
  const int tx = threadIdx.x & 31, ty = threadIdx.x >> 5;   // 32 x 8
#pragma unroll
  for (int i = 0; i < 32; i += 8)
    tile[ty + i][tx] = in[(size_t)(kb + ty + i) * N + nb + tx];
  __syncthreads();
#pragma unroll
  for (int i = 0; i < 32; i += 8)
    out[(size_t)(nb + ty + i) * K + kb + tx] = (__bf16)tile[tx][ty + i];
}

__global__ void pack_bias_kernel(const float* __restrict__ bq,
                                 const float* __restrict__ bk,
                                 const float* __restrict__ bv,
                                 float* __restrict__ out) {
  int i = blockIdx.x * 256 + threadIdx.x;
  if (i >= 3 * Cc) return;
  out[i] = (i < Cc) ? bq[i] : (i < 2 * Cc) ? bk[i - Cc] : bv[i - 2 * Cc];
}

// ---------------------------------------------------------------------------
// LayerNorm (one 256-thread block per row of C=1024) -> bf16 output
// ---------------------------------------------------------------------------

__global__ __launch_bounds__(256) void layernorm_to_bf16(
    const float* __restrict__ x, const float* __restrict__ g,
    const float* __restrict__ bta, __bf16* __restrict__ out) {
  __shared__ float red[256];
  const int row = blockIdx.x, tid = threadIdx.x;
  const float* xr = x + (size_t)row * Cc;
  float vals[4], s = 0.0f;
#pragma unroll
  for (int i = 0; i < 4; ++i) { vals[i] = xr[tid + i * 256]; s += vals[i]; }
  red[tid] = s; __syncthreads();
  for (int off = 128; off > 0; off >>= 1) {
    if (tid < off) red[tid] += red[tid + off];
    __syncthreads();
  }
  const float mean = red[0] * (1.0f / Cc);
  __syncthreads();
  float v = 0.0f;
#pragma unroll
  for (int i = 0; i < 4; ++i) { float d = vals[i] - mean; v += d * d; }
  red[tid] = v; __syncthreads();
  for (int off = 128; off > 0; off >>= 1) {
    if (tid < off) red[tid] += red[tid + off];
    __syncthreads();
  }
  const float rstd = rsqrtf(red[0] * (1.0f / Cc) + 1e-5f);
  __bf16* orow = out + (size_t)row * Cc;
#pragma unroll
  for (int i = 0; i < 4; ++i) {
    int c = tid + i * 256;
    orow[c] = (__bf16)((vals[i] - mean) * rstd * g[c] + bta[c]);
  }
}

// ---------------------------------------------------------------------------
// Tiled bf16 WMMA GEMM: C[M,N] = A[M,K] @ B[K,N] with B given TRANSPOSED as
// Bt[N][K]. 256 threads = 8 waves; block tile 128x128, K-step 32; wave tile
// 64x32. Double-buffered LDS; staging via TDM (tensor_load_to_lds) when
// available, else register-staged 16B copies. One barrier per K-step.
// mode 0: QKV  -> scatter bf16 to q/k/v [B,H,T,HS], q scaled by 1/sqrt(HS)
// mode 1: proj -> outF = resid + acc + bias (fp32)
// mode 2: FF1  -> outB0 = relu(acc + bias) (bf16)
// mode 3: FF2  -> outF = resid + acc + bias (fp32, final output)
// ---------------------------------------------------------------------------

__global__ __launch_bounds__(256) void gemm_bf16_wmma(
    const __bf16* __restrict__ A, const __bf16* __restrict__ Bt,
    int M, int N, int K,
    const float* __restrict__ bias, const float* __restrict__ resid,
    float* __restrict__ outF,
    __bf16* __restrict__ outB0, __bf16* __restrict__ outB1,
    __bf16* __restrict__ outB2, int mode) {
  __shared__ __bf16 As[2][128 * 32];   // [m][k] row-major, 64B rows
  __shared__ __bf16 Bs[2][128 * 32];   // [n][k] row-major, 64B rows

  const int tid = threadIdx.x;
  const int lane = tid & 31, lane16 = lane & 15, half = lane >> 4;
  const int wave = tid >> 5;
  const int m0 = (wave & 1) * 64;       // 2 waves along M
  const int n0 = (wave >> 1) * 32;      // 4 waves along N
  const int mBase = blockIdx.y * 128;
  const int nBase = blockIdx.x * 128;
  (void)M;

  v8f acc[4][2];
#pragma unroll
  for (int mi = 0; mi < 4; ++mi)
#pragma unroll
    for (int ni = 0; ni < 2; ++ni) acc[mi][ni] = v8f_zero();

  const int nk = K >> 5;

#if GEMM_USE_TDM
  // ---- TDM staging: wave 0 drives two DMA descriptors per K-step ----
  const __bf16* aTile = A  + (size_t)mBase * K;
  const __bf16* bTile = Bt + (size_t)nBase * K;
  const unsigned ldsA0 = (unsigned)(uintptr_t)&As[0][0];
  const unsigned ldsA1 = (unsigned)(uintptr_t)&As[1][0];
  const unsigned ldsB0 = (unsigned)(uintptr_t)&Bs[0][0];
  const unsigned ldsB1 = (unsigned)(uintptr_t)&Bs[1][0];

  if (tid < 32) {                       // prologue: tile 0 -> buffer 0
    tdm_load_tile(aTile, ldsA0, K);
    tdm_load_tile(bTile, ldsB0, K);
  }
  for (int kt = 0; kt < nk; ++kt) {
    const int buf = kt & 1;
    if (tid < 32) wait_tensorcnt0();    // current buffer's DMAs complete
    __syncthreads();                    // publish buffer to all waves
    if (kt + 1 < nk && tid < 32) {      // kick next tile into other buffer
      tdm_load_tile(aTile + ((size_t)(kt + 1) << 5), buf ? ldsA0 : ldsA1, K);
      tdm_load_tile(bTile + ((size_t)(kt + 1) << 5), buf ? ldsB0 : ldsB1, K);
    }
    v16bf af[4], bfr[2];
#pragma unroll
    for (int mi = 0; mi < 4; ++mi)
      af[mi] = load_a_frag(&As[buf][(m0 + mi * 16 + lane16) * 32], half);
#pragma unroll
    for (int ni = 0; ni < 2; ++ni)
      bfr[ni] = load_b_frag(&Bs[buf][(n0 + ni * 16 + lane16) * 32], half);
#pragma unroll
    for (int mi = 0; mi < 4; ++mi)
#pragma unroll
      for (int ni = 0; ni < 2; ++ni)
        acc[mi][ni] = wmma_bf16(af[mi], bfr[ni], acc[mi][ni]);
  }
#else
  // ---- Fallback: register-staged double buffering ----
  const int rowA0 = tid >> 2,          kOffA0 = (tid & 3) * 8;
  const int rowA1 = (tid + 256) >> 2,  kOffA1 = ((tid + 256) & 3) * 8;

  const __bf16* aSrc0 = A  + (size_t)(mBase + rowA0) * K + kOffA0;
  const __bf16* aSrc1 = A  + (size_t)(mBase + rowA1) * K + kOffA1;
  const __bf16* bSrc0 = Bt + (size_t)(nBase + rowA0) * K + kOffA0;
  const __bf16* bSrc1 = Bt + (size_t)(nBase + rowA1) * K + kOffA1;

  __bf16* aDst0 = &As[0][rowA0 * 32 + kOffA0];
  __bf16* aDst1 = &As[0][rowA1 * 32 + kOffA1];
  __bf16* bDst0 = &Bs[0][rowA0 * 32 + kOffA0];
  __bf16* bDst1 = &Bs[0][rowA1 * 32 + kOffA1];

  v8bf aReg0, aReg1, bReg0, bReg1;
  aReg0 = *(const v8bf*)(aSrc0);
  aReg1 = *(const v8bf*)(aSrc1);
  bReg0 = *(const v8bf*)(bSrc0);
  bReg1 = *(const v8bf*)(bSrc1);
  *(v8bf*)aDst0 = aReg0;  *(v8bf*)aDst1 = aReg1;
  *(v8bf*)bDst0 = bReg0;  *(v8bf*)bDst1 = bReg1;

  for (int kt = 0; kt < nk; ++kt) {
    __syncthreads();
    const int buf = kt & 1;
    const int kNext = (kt + 1) << 5;

    if (kt + 1 < nk) {
      aReg0 = *(const v8bf*)(aSrc0 + kNext);
      aReg1 = *(const v8bf*)(aSrc1 + kNext);
      bReg0 = *(const v8bf*)(bSrc0 + kNext);
      bReg1 = *(const v8bf*)(bSrc1 + kNext);
    }
    if (kt + 2 < nk) {
      const int kPf = (kt + 2) << 5;
      __builtin_prefetch(aSrc0 + kPf, 0, 3);
      __builtin_prefetch(bSrc0 + kPf, 0, 3);
    }

    v16bf af[4], bfr[2];
#pragma unroll
    for (int mi = 0; mi < 4; ++mi)
      af[mi] = load_a_frag(&As[buf][(m0 + mi * 16 + lane16) * 32], half);
#pragma unroll
    for (int ni = 0; ni < 2; ++ni)
      bfr[ni] = load_b_frag(&Bs[buf][(n0 + ni * 16 + lane16) * 32], half);
#pragma unroll
    for (int mi = 0; mi < 4; ++mi)
#pragma unroll
      for (int ni = 0; ni < 2; ++ni)
        acc[mi][ni] = wmma_bf16(af[mi], bfr[ni], acc[mi][ni]);

    if (kt + 1 < nk) {
      const int boff = (buf ^ 1) * (128 * 32);
      *(v8bf*)(aDst0 + boff) = aReg0;  *(v8bf*)(aDst1 + boff) = aReg1;
      *(v8bf*)(bDst0 + boff) = bReg0;  *(v8bf*)(bDst1 + boff) = bReg1;
    }
  }
#endif

  // Epilogue. C/D layout: lane -> col N = lane%16; vgpr r -> row M = r + 8*(lane/16)
#pragma unroll
  for (int mi = 0; mi < 4; ++mi) {
#pragma unroll
    for (int ni = 0; ni < 2; ++ni) {
      const int col = nBase + n0 + ni * 16 + lane16;
      const float bv = bias ? bias[col] : 0.0f;
#pragma unroll
      for (int r = 0; r < 8; ++r) {
        const int rowG = mBase + m0 + mi * 16 + r + 8 * half;
        float val = acc[mi][ni][r] + bv;
        if (mode == 0) {
          const int which = col >> 10, hh = (col >> 6) & 15, dd = col & 63;
          const int b = rowG >> 11, t = rowG & (Ts - 1);
          const size_t dst = (((size_t)(b * Hh + hh)) * Ts + t) * HSz + dd;
          if (which == 0)      outB0[dst] = (__bf16)(val * 0.125f); // pre-scale Q
          else if (which == 1) outB1[dst] = (__bf16)val;
          else                 outB2[dst] = (__bf16)val;
        } else if (mode == 2) {
          outB0[(size_t)rowG * N + col] = (__bf16)fmaxf(val, 0.0f);
        } else { // 1 or 3: residual add, fp32 out
          outF[(size_t)rowG * N + col] = resid[(size_t)rowG * N + col] + val;
        }
      }
    }
  }
}

// ---------------------------------------------------------------------------
// Causal flash attention. Grid: (T/64, B*H); 128 threads = 4 waves.
// Wave w owns 16 query rows; iterate 32-wide key tiles with online softmax.
// QK^T: A = Q rows (kept in regs), B = K rows (K is [s][d] = exactly the
// k-major layout the B-fragment wants). PV: P goes through LDS (C-layout ->
// A-layout), V staged transposed [d][s] in LDS.
// ---------------------------------------------------------------------------

__global__ __launch_bounds__(128) void attn_flash_wmma(
    const __bf16* __restrict__ Q, const __bf16* __restrict__ K,
    const __bf16* __restrict__ V, __bf16* __restrict__ O) {
  __shared__ __bf16 VT[64][32];      // [d][s] for current key tile
  __shared__ __bf16 Pt[4][16][32];   // per-wave probability tile

  const int tid = threadIdx.x;
  const int lane = tid & 31, lane16 = lane & 15, half = lane >> 4;
  const int wave = tid >> 5;
  const int bh = blockIdx.y;
  const int qBlock = blockIdx.x;
  const size_t headOff = (size_t)bh * Ts * HSz;
  const __bf16* Qh = Q + headOff;
  const __bf16* Kh = K + headOff;
  const __bf16* Vh = V + headOff;
  const int qRow0 = qBlock * 64 + wave * 16;

  // Q fragments live in registers for the whole pass (Q pre-scaled by 1/8)
  v16bf qf[2];
  {
    const __bf16* qrow = Qh + (size_t)(qRow0 + lane16) * HSz;
    qf[0] = load_a_frag(qrow, half);
    qf[1] = load_a_frag(qrow + 32, half);
  }

  v8f o_acc[4];
#pragma unroll
  for (int dt = 0; dt < 4; ++dt) o_acc[dt] = v8f_zero();
  float m_i[8], l_i[8];
#pragma unroll
  for (int r = 0; r < 8; ++r) { m_i[r] = -1e30f; l_i[r] = 0.0f; }

  const int sEnd = qBlock * 64 + 63;
  for (int s0 = 0; s0 <= sEnd; s0 += 32) {
    // Prefetch next tile's K/V rows (rows are 128B = one line each)
    if (s0 + 32 <= sEnd) {
      const int pr = s0 + 32 + (tid & 31);
      __builtin_prefetch(Kh + (size_t)pr * HSz, 0, 3);
      __builtin_prefetch(Vh + (size_t)pr * HSz, 0, 3);
    }
    // Stage V tile transposed (coalesced global read, scattered LDS write)
#pragma unroll
    for (int i = 0; i < 16; ++i) {
      int idx = i * 128 + tid;
      int ss = idx >> 6, dd = idx & 63;
      VT[dd][ss] = Vh[(size_t)(s0 + ss) * HSz + dd];
    }
    __syncthreads();

    // S = Q @ K^T : two 16-col tiles, K-dim 64 = 2 WMMAs each
    v8f sacc[2];
    sacc[0] = v8f_zero(); sacc[1] = v8f_zero();
#pragma unroll
    for (int nt = 0; nt < 2; ++nt) {
      const __bf16* krow = Kh + (size_t)(s0 + nt * 16 + lane16) * HSz;
#pragma unroll
      for (int j = 0; j < 2; ++j)
        sacc[nt] = wmma_bf16(qf[j], load_b_frag(krow + j * 32, half), sacc[nt]);
    }

    // Causal mask + online softmax (row stats reduced across the 16-lane half)
#pragma unroll
    for (int r = 0; r < 8; ++r) {
      const int qg = qRow0 + r + 8 * half;
      float v0 = (s0 + lane16 > qg)      ? -1e30f : sacc[0][r];
      float v1 = (s0 + 16 + lane16 > qg) ? -1e30f : sacc[1][r];
      float mx = rmax16(fmaxf(v0, v1));
      float mnew = fmaxf(m_i[r], mx);
      float alpha = __expf(m_i[r] - mnew);
      float p0 = __expf(v0 - mnew);
      float p1 = __expf(v1 - mnew);
      float sum = rsum16(p0 + p1);
      l_i[r] = l_i[r] * alpha + sum;
      m_i[r] = mnew;
#pragma unroll
      for (int dt = 0; dt < 4; ++dt) o_acc[dt][r] *= alpha;
      Pt[wave][r + 8 * half][lane16]      = (__bf16)p0;
      Pt[wave][r + 8 * half][16 + lane16] = (__bf16)p1;
    }
    __syncthreads();

    // O += P @ V  (A-frag from Pt, B-frags from transposed V tile)
    v16bf pa = load_a_frag(&Pt[wave][lane16][0], half);
#pragma unroll
    for (int dt = 0; dt < 4; ++dt) {
      v16bf vb = load_b_frag(&VT[dt * 16 + lane16][0], half);
      o_acc[dt] = wmma_bf16(pa, vb, o_acc[dt]);
    }
    __syncthreads();  // protect VT/Pt before next iteration's staging
  }

  // Normalize and write O as bf16 [B, T, H*HS] (head-concat layout for proj)
  const int b = bh >> 4, h = bh & 15;
  float inv[8];
#pragma unroll
  for (int r = 0; r < 8; ++r) inv[r] = 1.0f / l_i[r];
#pragma unroll
  for (int dt = 0; dt < 4; ++dt) {
#pragma unroll
    for (int r = 0; r < 8; ++r) {
      const int t = qRow0 + r + 8 * half;
      const int col = h * HSz + dt * 16 + lane16;
      O[((size_t)(b * Ts + t)) * Cc + col] = (__bf16)(o_acc[dt][r] * inv[r]);
    }
  }
}

// ---------------------------------------------------------------------------
// Host launcher
// ---------------------------------------------------------------------------

extern "C" void kernel_launch(void* const* d_in, const int* in_sizes, int n_in,
                              void* d_out, int out_size, void* d_ws, size_t ws_size,
                              hipStream_t stream) {
  (void)in_sizes; (void)n_in; (void)out_size; (void)ws_size;

  const float* x     = (const float*)d_in[0];
  const float* Wq    = (const float*)d_in[1];
  const float* bq    = (const float*)d_in[2];
  const float* Wk    = (const float*)d_in[3];
  const float* bk    = (const float*)d_in[4];
  const float* Wv    = (const float*)d_in[5];
  const float* bv    = (const float*)d_in[6];
  const float* Wproj = (const float*)d_in[7];
  const float* bproj = (const float*)d_in[8];
  const float* W1    = (const float*)d_in[9];
  const float* b1    = (const float*)d_in[10];
  const float* W2    = (const float*)d_in[11];
  const float* b2    = (const float*)d_in[12];
  const float* ln1_g = (const float*)d_in[13];
  const float* ln1_b = (const float*)d_in[14];
  const float* ln2_g = (const float*)d_in[15];
  const float* ln2_b = (const float*)d_in[16];
  float* out = (float*)d_out;

  // Workspace carve-up (256B aligned)
  char* w = (char*)d_ws;
  auto take = [&](size_t bytes) {
    char* p = w;
    w += (bytes + 255) & ~(size_t)255;
    return p;
  };
  __bf16* hBf      = (__bf16*)take((size_t)Mrows * Cc * 2);
  __bf16* WqkvT    = (__bf16*)take((size_t)3 * Cc * Cc * 2);   // [3072][1024]
  float*  bqkv     = (float*) take((size_t)3 * Cc * 4);
  __bf16* qBf      = (__bf16*)take((size_t)Mrows * Cc * 2);
  __bf16* kBf      = (__bf16*)take((size_t)Mrows * Cc * 2);
  __bf16* vBf      = (__bf16*)take((size_t)Mrows * Cc * 2);
  __bf16* oBf      = (__bf16*)take((size_t)Mrows * Cc * 2);
  float*  x1       = (float*) take((size_t)Mrows * Cc * 4);
  __bf16* WprojT   = (__bf16*)take((size_t)Cc * Cc * 2);       // [1024][1024]
  __bf16* h2Bf     = (__bf16*)take((size_t)Mrows * Cc * 2);
  __bf16* W1T      = (__bf16*)take((size_t)4 * Cc * Cc * 2);   // [4096][1024]
  __bf16* ff1Bf    = (__bf16*)take((size_t)Mrows * 4 * Cc * 2);
  __bf16* W2T      = (__bf16*)take((size_t)4 * Cc * Cc * 2);   // [1024][4096]

  // 1) Repack weights/biases to transposed bf16 (once per call)
  {
    int n = 3 * Cc * Cc;
    pack_qkv_wT_kernel<<<(n + 255) / 256, 256, 0, stream>>>(Wq, Wk, Wv, WqkvT);
    pack_bias_kernel<<<(3 * Cc + 255) / 256, 256, 0, stream>>>(bq, bk, bv, bqkv);
    // in [K][N] fp32 -> out [N][K] bf16 ; grid (N/32, K/32)
    transpose_to_bf16<<<dim3(Cc / 32, Cc / 32), 256, 0, stream>>>(
        Wproj, WprojT, Cc, Cc);
    transpose_to_bf16<<<dim3(4 * Cc / 32, Cc / 32), 256, 0, stream>>>(
        W1, W1T, Cc, 4 * Cc);
    transpose_to_bf16<<<dim3(Cc / 32, 4 * Cc / 32), 256, 0, stream>>>(
        W2, W2T, 4 * Cc, Cc);
  }

  // 2) LN1 -> bf16 activations
  layernorm_to_bf16<<<Mrows, 256, 0, stream>>>(x, ln1_g, ln1_b, hBf);

  // 3) Fused QKV GEMM: [8192,1024] @ [1024,3072], scatter to per-head q/k/v
  gemm_bf16_wmma<<<dim3(3 * Cc / 128, Mrows / 128), 256, 0, stream>>>(
      hBf, WqkvT, Mrows, 3 * Cc, Cc, bqkv, nullptr, nullptr,
      qBf, kBf, vBf, /*mode=*/0);

  // 4) Causal flash attention (grid: 32 query blocks x 64 heads)
  attn_flash_wmma<<<dim3(Ts / 64, Bb * Hh), 128, 0, stream>>>(qBf, kBf, vBf, oBf);

  // 5) Output projection + residual -> x1 (fp32)
  gemm_bf16_wmma<<<dim3(Cc / 128, Mrows / 128), 256, 0, stream>>>(
      oBf, WprojT, Mrows, Cc, Cc, bproj, x, x1,
      nullptr, nullptr, nullptr, /*mode=*/1);

  // 6) LN2 -> bf16
  layernorm_to_bf16<<<Mrows, 256, 0, stream>>>(x1, ln2_g, ln2_b, h2Bf);

  // 7) FFN up + ReLU -> bf16
  gemm_bf16_wmma<<<dim3(4 * Cc / 128, Mrows / 128), 256, 0, stream>>>(
      h2Bf, W1T, Mrows, 4 * Cc, Cc, b1, nullptr, nullptr,
      ff1Bf, nullptr, nullptr, /*mode=*/2);

  // 8) FFN down + residual -> final fp32 output
  gemm_bf16_wmma<<<dim3(Cc / 128, Mrows / 128), 256, 0, stream>>>(
      ff1Bf, W2T, Mrows, Cc, 4 * Cc, b2, x1, out,
      nullptr, nullptr, nullptr, /*mode=*/3);
}
